// YiJingAttention_49108656062775
// MI455X (gfx1250) — matrix operand order repaired and checked by reading, not compiled, not run on hardware
//
#include <hip/hip_runtime.h>

#define DEVI __device__ __forceinline__

typedef unsigned short u16t;
typedef __attribute__((ext_vector_type(4)))  unsigned int v4u;
typedef __attribute__((ext_vector_type(8)))  unsigned int v8u;
typedef __attribute__((ext_vector_type(16))) __bf16      v16bf;
typedef __attribute__((ext_vector_type(8)))  float       v8f;

constexpr int B_   = 2;
constexpr int T_   = 2048;
constexpr int DM   = 1024;
constexpr int NH   = 8;
constexpr int NKV  = 4;
constexpr int HD   = 128;
constexpr int M_   = B_ * T_;     // 4096 token rows
constexpr int WIN  = 512;

// ---------- bf16 helpers (manual RNE; storage is ushort, WMMA frags via bit_cast) ----------
DEVI u16t f2bf(float f) {
  unsigned u = __builtin_bit_cast(unsigned, f);
  u += 0x7fffu + ((u >> 16) & 1u);
  return (u16t)(u >> 16);
}
DEVI float bf2f(u16t h) {
  unsigned u = ((unsigned)h) << 16;
  return __builtin_bit_cast(float, u);
}
DEVI v16bf pack16(v4u lo, v4u hi) {
  v8u r;
  r[0]=lo[0]; r[1]=lo[1]; r[2]=lo[2]; r[3]=lo[3];
  r[4]=hi[0]; r[5]=hi[1]; r[6]=hi[2]; r[7]=hi[3];
  return __builtin_bit_cast(v16bf, r);
}

// ---------- elementwise f32 -> bf16 ----------
__global__ __launch_bounds__(256)
void yj_cvt_bf16(const float* __restrict__ src, u16t* __restrict__ dst, int n) {
  int i = blockIdx.x * 256 + threadIdx.x;
  if (i < n) dst[i] = f2bf(src[i]);
}

// ---------- W [K x N] f32 -> Wt [N x K] bf16 (so B-fragments are contiguous per lane) ----------
__global__ __launch_bounds__(256)
void yj_wtrans(const float* __restrict__ src, u16t* __restrict__ dst, int K, int N) {
  int i = blockIdx.x * 256 + threadIdx.x;
  if (i >= K * N) return;
  int k = i % K;
  int n = i / K;
  dst[n * K + k] = f2bf(src[k * N + n]);
}

// ---------- bf16 GEMM: C[M x N] = A[M x K] * Bt[N x K]^T, f32 accumulate via WMMA ----------
// block = 128 threads (4 waves); wave computes 16(M) x 64(N) with 4 accumulators.
// Register double-buffered K-loop; loads are issued in WMMA-consumption order
// (A first, then B0..B3) so the backend can retire s_wait_loadcnt incrementally
// instead of draining to 0 before the first WMMA.
__global__ __launch_bounds__(128)
void yj_gemm_bf16(const u16t* __restrict__ A, const u16t* __restrict__ Bt,
                  float* __restrict__ C, int M, int N, int K) {
  const int lane  = threadIdx.x & 31;
  const int wave  = threadIdx.x >> 5;
  const int m0    = blockIdx.y * 16;
  const int nbase = blockIdx.x * 256 + wave * 64;
  const int nloc  = lane & 15;
  const int half  = lane >> 4;
  const int row   = m0 + nloc;

  v8f acc[4] = {};
  const u16t* arow = A + row * K;
  const u16t* bcol[4];
#pragma unroll
  for (int t = 0; t < 4; ++t)
    bcol[t] = Bt + (nbase + t * 16 + nloc) * K + half * 16;

  // prologue: load K-step 0 fragments (A first, then B in consumption order)
  v4u alo = *(const v4u*)(arow + half * 16);
  v4u ahi = *(const v4u*)(arow + 8 + half * 16);
  v8u breg[4];
#pragma unroll
  for (int t = 0; t < 4; ++t) breg[t] = *(const v8u*)(bcol[t]);

  for (int kk = 0; kk < K - 32; kk += 32) {
    const int kn = kk + 32;
    __builtin_prefetch(arow + kn + 160, 0, 3);  // near-cache global_prefetch
    // issue next K-step loads first, A before B (matches consumption order) ...
    v4u nlo = *(const v4u*)(arow + kn + half * 16);
    v4u nhi = *(const v4u*)(arow + kn + 8 + half * 16);
    v8u nb0 = *(const v8u*)(bcol[0] + kn);
    v8u nb1 = *(const v8u*)(bcol[1] + kn);
    v8u nb2 = *(const v8u*)(bcol[2] + kn);
    v8u nb3 = *(const v8u*)(bcol[3] + kn);
    // ... then compute on current buffers (loads overlap WMMA)
    v16bf af = pack16(alo, ahi);
#pragma unroll
    for (int t = 0; t < 4; ++t) {
      v16bf bfr = __builtin_bit_cast(v16bf, breg[t]);
      acc[t] = __builtin_amdgcn_wmma_f32_16x16x32_bf16(
          false, af, false, bfr, (short)0, acc[t], false, false);
    }
    alo = nlo; ahi = nhi;
    breg[0] = nb0; breg[1] = nb1; breg[2] = nb2; breg[3] = nb3;
  }
  // epilogue: last K-step
  {
    v16bf af = pack16(alo, ahi);
#pragma unroll
    for (int t = 0; t < 4; ++t) {
      v16bf bfr = __builtin_bit_cast(v16bf, breg[t]);
      acc[t] = __builtin_amdgcn_wmma_f32_16x16x32_bf16(
          false, af, false, bfr, (short)0, acc[t], false, false);
    }
  }

#pragma unroll
  for (int t = 0; t < 4; ++t) {
    int col = nbase + t * 16 + nloc;
#pragma unroll
    for (int r = 0; r < 8; ++r) {
      int m = m0 + half * 8 + r;
      C[m * N + col] = acc[t][r];
    }
  }
}

// ---------- RoPE (f32) + quantize to bf16 for Q and K ----------
__global__ __launch_bounds__(256)
void yj_rope(const float* __restrict__ Qf, const float* __restrict__ Kf,
             u16t* __restrict__ Qb, u16t* __restrict__ Kb) {
  int tid = blockIdx.x * 256 + threadIdx.x;
  int d    = tid & 63;
  int head = (tid >> 6) % (NH + NKV);
  int rowg = tid / (64 * (NH + NKV));
  if (rowg >= M_) return;
  int t = rowg & (T_ - 1);

  // inv_freq = 10000^(-d/64) = exp(-(d/64)*ln(10000))
  float inv = __expf(-0.015625f * (float)d * 9.210340371976184f);
  float ang = (float)t * inv;
  float s, c;
  __sincosf(ang, &s, &c);

  if (head < NH) {
    const float* src = Qf + rowg * (NH * HD) + head * HD;
    u16t*        dst = Qb + rowg * (NH * HD) + head * HD;
    float x1 = src[d], x2 = src[d + 64];
    dst[d]      = f2bf(x1 * c - x2 * s);
    dst[d + 64] = f2bf(x2 * c + x1 * s);
  } else {
    int kvh = head - NH;
    const float* src = Kf + rowg * (NKV * HD) + kvh * HD;
    u16t*        dst = Kb + rowg * (NKV * HD) + kvh * HD;
    float x1 = src[d], x2 = src[d + 64];
    dst[d]      = f2bf(x1 * c - x2 * s);
    dst[d + 64] = f2bf(x2 * c + x1 * s);
  }
}

// ---------- V f32 [token x 512] -> Vt bf16 [(b,kvh,dim) x T] ----------
__global__ __launch_bounds__(256)
void yj_vtrans(const float* __restrict__ Vf, u16t* __restrict__ Vt) {
  int tid = blockIdx.x * 256 + threadIdx.x;
  if (tid >= B_ * NKV * HD * T_) return;
  int t = tid & (T_ - 1);
  int r = tid >> 11;          // / T_
  int d   = r & (HD - 1); r >>= 7;
  int kvh = r & (NKV - 1);
  int b   = r >> 2;
  Vt[((b * NKV + kvh) * HD + d) * T_ + t] =
      f2bf(Vf[(b * T_ + t) * (NKV * HD) + kvh * HD + d]);
}

// ---------- qp/kp geo projections (first 3 dims . dirs[h]) ----------
__global__ __launch_bounds__(256)
void yj_geoproj(const u16t* __restrict__ Qb, const u16t* __restrict__ Kb,
                float* __restrict__ qp, float* __restrict__ kp) {
  int tid = blockIdx.x * 256 + threadIdx.x;
  const int total = B_ * NH * T_;
  if (tid >= 2 * total) return;
  int which = (tid >= total);
  int i = which ? tid - total : tid;
  int t = i & (T_ - 1);
  int h = (i >> 11) & (NH - 1);
  int b = i >> 14;
  const float s3 = 0.5773502691896258f;  // 1/sqrt(3)
  float d0 = (h & 4) ? s3 : -s3;
  float d1 = (h & 2) ? s3 : -s3;
  float d2 = (h & 1) ? s3 : -s3;
  if (!which) {
    const u16t* p = Qb + (b * T_ + t) * (NH * HD) + h * HD;
    qp[i] = bf2f(p[0]) * d0 + bf2f(p[1]) * d1 + bf2f(p[2]) * d2;
  } else {
    const u16t* p = Kb + (b * T_ + t) * (NKV * HD) + (h >> 1) * HD;
    kp[i] = bf2f(p[0]) * d0 + bf2f(p[1]) * d1 + bf2f(p[2]) * d2;
  }
}

// ---------- sliding-window flash attention ----------
// block = 128 threads = 4 wave32 waves, each wave owns an independent 16-query
// tile (separate key-loop trip counts -> no block barrier; same-wave DS ops are
// in-order on CDNA5, so the P transpose only needs a compiler barrier).
__global__ __launch_bounds__(128)
void yj_attn(const u16t* __restrict__ Qb, const u16t* __restrict__ Kb,
             const u16t* __restrict__ Vt, const float* __restrict__ qp,
             const float* __restrict__ kp, const float* __restrict__ head_scales,
             u16t* __restrict__ AOb) {
  __shared__ u16t PldsAll[4][16 * 32];

  const int lane   = threadIdx.x & 31;
  const int wave   = threadIdx.x >> 5;
  u16t* Plds       = PldsAll[wave];
  const int t0     = blockIdx.x * 64 + wave * 16;
  const int h      = blockIdx.y;
  const int b      = blockIdx.z;
  const int kvh    = h >> 1;
  const int nloc   = lane & 15;
  const int half   = lane >> 4;
  const int base_m = half * 8;
  const float scale = 0.08838834764831845f;  // 1/sqrt(128)
  const float hs    = head_scales[h];

  // resident Q fragments: 4 chunks of K=32 along head dim
  const u16t* qrow = Qb + (b * T_ + t0 + nloc) * (NH * HD) + h * HD;
  v16bf aq[4];
#pragma unroll
  for (int c4 = 0; c4 < 4; ++c4) {
    v4u lo = *(const v4u*)(qrow + c4 * 32 + half * 16);
    v4u hi = *(const v4u*)(qrow + c4 * 32 + 8 + half * 16);
    aq[c4] = pack16(lo, hi);
  }
  float qpv[8];
#pragma unroll
  for (int r = 0; r < 8; ++r)
    qpv[r] = qp[(b * NH + h) * T_ + t0 + base_m + r] * hs;

  v8f o[8] = {};
  float mrow[8], lrow[8];
#pragma unroll
  for (int r = 0; r < 8; ++r) { mrow[r] = -1e30f; lrow[r] = 0.0f; }

  int lo_key = t0 - (WIN - 1);
  if (lo_key < 0) lo_key = 0;
  lo_key &= ~31;

  for (int s0 = lo_key; s0 <= t0 + 15; s0 += 32) {
    // ---- S = Q * K^T for 32 keys (two 16-wide tiles) ----
    v8f S0 = {}, S1 = {};
    {
      const u16t* krow0 = Kb + (b * T_ + s0 + nloc) * (NKV * HD) + kvh * HD;
      const u16t* krow1 = krow0 + 16 * (NKV * HD);
#pragma unroll
      for (int c4 = 0; c4 < 4; ++c4) {
        v8u b0 = *(const v8u*)(krow0 + c4 * 32 + half * 16);
        v8u b1 = *(const v8u*)(krow1 + c4 * 32 + half * 16);
        v16bf bk0 = __builtin_bit_cast(v16bf, b0);
        v16bf bk1 = __builtin_bit_cast(v16bf, b1);
        S0 = __builtin_amdgcn_wmma_f32_16x16x32_bf16(false, aq[c4], false, bk0,
                                                     (short)0, S0, false, false);
        S1 = __builtin_amdgcn_wmma_f32_16x16x32_bf16(false, aq[c4], false, bk1,
                                                     (short)0, S1, false, false);
      }
    }

    // ---- bias + window mask + online softmax (rows live in 16-lane halves) ----
    const int j0 = s0 + nloc;
    const int j1 = s0 + 16 + nloc;
    const float kp0 = kp[(b * NH + h) * T_ + j0];
    const float kp1 = kp[(b * NH + h) * T_ + j1];
    float p0[8], p1[8];
#pragma unroll
    for (int r = 0; r < 8; ++r) {
      const int i = t0 + base_m + r;
      float s0v = (j0 <= i && j0 >= i - (WIN - 1)) ? S0[r] * scale + qpv[r] * kp0 : -1e30f;
      float s1v = (j1 <= i && j1 >= i - (WIN - 1)) ? S1[r] * scale + qpv[r] * kp1 : -1e30f;
      float mx = fmaxf(s0v, s1v);
      for (int off = 8; off; off >>= 1) mx = fmaxf(mx, __shfl_xor(mx, off, 16));
      float mnew = fmaxf(mrow[r], mx);
      float sc = __expf(mrow[r] - mnew);
      float e0 = (s0v < -1e29f) ? 0.0f : __expf(s0v - mnew);
      float e1 = (s1v < -1e29f) ? 0.0f : __expf(s1v - mnew);
      float rs = e0 + e1;
      for (int off = 8; off; off >>= 1) rs += __shfl_xor(rs, off, 16);
      lrow[r] = lrow[r] * sc + rs;
      mrow[r] = mnew;
      p0[r] = e0;
      p1[r] = e1;
#pragma unroll
      for (int c8 = 0; c8 < 8; ++c8) o[c8][r] *= sc;
    }

    // ---- transpose P (C-layout -> A-layout) through this wave's LDS tile ----
    // Same-wave DS ops execute in order on CDNA5; only stop compiler reordering.
    asm volatile("" ::: "memory");
#pragma unroll
    for (int r = 0; r < 8; ++r) {
      Plds[(base_m + r) * 32 + nloc]      = f2bf(p0[r]);
      Plds[(base_m + r) * 32 + 16 + nloc] = f2bf(p1[r]);
    }
    asm volatile("" ::: "memory");
    v4u plo = *(const v4u*)(&Plds[nloc * 32 + half * 16]);
    v4u phi = *(const v4u*)(&Plds[nloc * 32 + 8 + half * 16]);
    v16bf ap = pack16(plo, phi);

    // ---- O += P * V  (V pre-transposed: contiguous keys per lane) ----
#pragma unroll
    for (int c8 = 0; c8 < 8; ++c8) {
      const u16t* vcol =
          Vt + ((b * NKV + kvh) * HD + c8 * 16 + nloc) * T_ + s0 + half * 16;
      v8u vr = *(const v8u*)vcol;
      v16bf bv = __builtin_bit_cast(v16bf, vr);
      o[c8] = __builtin_amdgcn_wmma_f32_16x16x32_bf16(false, ap, false, bv,
                                                      (short)0, o[c8], false, false);
    }
  }

  // ---- normalize + store bf16 for the output projection GEMM ----
  float rl[8];
#pragma unroll
  for (int r = 0; r < 8; ++r) rl[r] = 1.0f / lrow[r];
#pragma unroll
  for (int c8 = 0; c8 < 8; ++c8)
#pragma unroll
    for (int r = 0; r < 8; ++r)
      AOb[(b * T_ + t0 + base_m + r) * (NH * HD) + h * HD + c8 * 16 + nloc] =
          f2bf(o[c8][r] * rl[r]);
}

// =====================================================================
extern "C" void kernel_launch(void* const* d_in, const int* in_sizes, int n_in,
                              void* d_out, int out_size, void* d_ws, size_t ws_size,
                              hipStream_t stream) {
  const float* x   = (const float*)d_in[0];
  const float* Wq  = (const float*)d_in[1];
  const float* Wk  = (const float*)d_in[2];
  const float* Wv  = (const float*)d_in[3];
  const float* Wo  = (const float*)d_in[4];
  const float* hsc = (const float*)d_in[5];

  char* ws = (char*)d_ws;
  const size_t MB = 1024 * 1024;
  u16t*  xb  = (u16t*)(ws + 0);            // 8 MB  : x bf16        [4096 x 1024]
  u16t*  WqT = (u16t*)(ws + 8  * MB);      // 2 MB  : Wq^T bf16     [1024 x 1024]
  u16t*  WkT = (u16t*)(ws + 10 * MB);      // 1 MB  : Wk^T bf16     [512 x 1024]
  u16t*  WvT = (u16t*)(ws + 11 * MB);      // 1 MB  : Wv^T bf16     [512 x 1024]
  u16t*  WoT = (u16t*)(ws + 12 * MB);      // 2 MB  : Wo^T bf16     [1024 x 1024]
  float* Qf  = (float*)(ws + 14 * MB);     // 16 MB : Q f32         [4096 x 1024]
  float* Kf  = (float*)(ws + 30 * MB);     // 8 MB  : K f32         [4096 x 512]
  float* Vf  = (float*)(ws + 38 * MB);     // 8 MB  : V f32         [4096 x 512]
  u16t*  Qb  = (u16t*)(ws + 46 * MB);      // 8 MB  : roped Q bf16
  u16t*  Kb  = (u16t*)(ws + 54 * MB);      // 4 MB  : roped K bf16
  u16t*  Vt  = (u16t*)(ws + 58 * MB);      // 4 MB  : V^T bf16      [(b,kvh,d) x T]
  float* qpB = (float*)(ws + 62 * MB);     // 128 KB: qp [B,H,T]
  float* kpB = (float*)(ws + 62 * MB + 128 * 1024);  // 128 KB: kp
  u16t*  AOb = (u16t*)(ws + 63 * MB);      // 8 MB  : attn out bf16 [4096 x 1024]

  // 1) casts / transposed-weight casts
  yj_cvt_bf16<<<(M_ * DM) / 256, 256, 0, stream>>>(x, xb, M_ * DM);
  yj_wtrans<<<(1024 * 1024) / 256, 256, 0, stream>>>(Wq, WqT, 1024, 1024);
  yj_wtrans<<<(1024 * 512) / 256, 256, 0, stream>>>(Wk, WkT, 1024, 512);
  yj_wtrans<<<(1024 * 512) / 256, 256, 0, stream>>>(Wv, WvT, 1024, 512);
  yj_wtrans<<<(1024 * 1024) / 256, 256, 0, stream>>>(Wo, WoT, 1024, 1024);

  // 2) QKV projections (WMMA bf16 -> f32)
  yj_gemm_bf16<<<dim3(1024 / 256, M_ / 16), 128, 0, stream>>>(xb, WqT, Qf, M_, 1024, 1024);
  yj_gemm_bf16<<<dim3(512 / 256,  M_ / 16), 128, 0, stream>>>(xb, WkT, Kf, M_, 512, 1024);
  yj_gemm_bf16<<<dim3(512 / 256,  M_ / 16), 128, 0, stream>>>(xb, WvT, Vf, M_, 512, 1024);

  // 3) RoPE + quantize, V transpose, geo projections
  yj_rope<<<(M_ * (NH + NKV) * 64) / 256, 256, 0, stream>>>(Qf, Kf, Qb, Kb);
  yj_vtrans<<<(B_ * NKV * HD * T_) / 256, 256, 0, stream>>>(Vf, Vt);
  yj_geoproj<<<(2 * B_ * NH * T_) / 256, 256, 0, stream>>>(Qb, Kb, qpB, kpB);

  // 4) sliding-window flash attention (WMMA QK^T and PV)
  yj_attn<<<dim3(T_ / 64, NH, B_), 128, 0, stream>>>(Qb, Kb, Vt, qpB, kpB, hsc, AOb);

  // 5) output projection -> f32 d_out
  yj_gemm_bf16<<<dim3(1024 / 256, M_ / 16), 128, 0, stream>>>(AOb, WoT, (float*)d_out, M_, 1024, 1024);
}
